// RNNRegressor_86861418595088
// MI455X (gfx1250) — matrix operand order repaired and checked by reading, not compiled, not run on hardware
//
#include <hip/hip_runtime.h>
#include <hip/hip_bf16.h>

#define LL 15
#define TT 20
#define HH 64
#define ROWS 32
#define BB 8192
#define NBLK (BB / ROWS)
#define PACK_PER_LAYER 16384  /* ushorts: 32 fragments * 512 */

typedef __attribute__((ext_vector_type(16))) __bf16 v16bf;
typedef __attribute__((ext_vector_type(8)))  float  v8f;

union BfFrag { v16bf v; uint4 u[2]; };

__device__ __forceinline__ unsigned short f2bf(float f) {
    unsigned int u = __builtin_bit_cast(unsigned int, f);
    u += 0x7FFFu + ((u >> 16) & 1u);   // round-to-nearest-even
    return (unsigned short)(u >> 16);
}
__device__ __forceinline__ float bf2f(unsigned short s) {
    unsigned int u = ((unsigned int)s) << 16;
    return __builtin_bit_cast(float, u);
}

// Branch-free sigmoid/tanh: raw v_exp_f32 + v_rcp_f32, correct saturation at +/-inf.
__device__ __forceinline__ float sigf(float x) {
    return __builtin_amdgcn_rcpf(1.0f + __expf(-x));
}
__device__ __forceinline__ float tanh_fast(float x) {
    return 1.0f - 2.0f * __builtin_amdgcn_rcpf(1.0f + __expf(2.0f * x));
}

// A-fragment (16-bit, 16x32): lane L -> M = L%16, g = L/16; halves i:
// K(i) = 16*(i/8) + 8*g + (i%8)  ==> two contiguous 8-half (16B) runs.
__device__ __forceinline__ v16bf load_afrag(const unsigned short* tile, int m, int kt, int lane) {
    int row = 16 * m + (lane & 15);
    int g   = lane >> 4;
    const unsigned short* p = tile + row * HH + 32 * kt + 8 * g;
    BfFrag f;
    f.u[0] = *(const uint4*)(p);
    f.u[1] = *(const uint4*)(p + 16);
    return f.v;
}

// B-fragment from pre-packed weights: per-lane 32B contiguous (coalesced).
__device__ __forceinline__ v16bf load_bfrag(const unsigned short* pack, int frag, int lane) {
    const unsigned short* p = pack + frag * 512 + lane * 16;
    BfFrag f;
    f.u[0] = *(const uint4*)(p);
    f.u[1] = *(const uint4*)(p + 8);
    return f.v;
}

__device__ __forceinline__ v8f wmma_bf(v16bf a, v16bf b, v8f c) {
    return __builtin_amdgcn_wmma_f32_16x16x32_bf16(false, a, false, b, (short)0, c, false, false);
}

// LSTM cell nonlinearity on 4 gate fragments; updates c (regs), writes h (bf16) to LDS tile.
__device__ __forceinline__ void cell_update(const v8f* acc, float* c8, unsigned short* hdst,
                                            int m, int cg, int g, int ln) {
#pragma unroll
    for (int j = 0; j < 8; ++j) {
        float iv = sigf(acc[0][j]);
        float fv = sigf(acc[1][j]);
        float gv = tanh_fast(acc[2][j]);
        float ov = sigf(acc[3][j]);
        float cn = fv * c8[j] + iv * gv;
        c8[j] = cn;
        float h = ov * tanh_fast(cn);
        int row = 16 * m + j + 8 * g;
        hdst[row * HH + 16 * cg + ln] = f2bf(h);
    }
}

// ---------------- prep: pack weights into WMMA B-fragment order, fuse biases --------------
__global__ void lstm_prep(const float* __restrict__ Whh0,
                          const float* __restrict__ bih0,
                          const float* __restrict__ bhh0,
                          const float* __restrict__ Wih,
                          const float* __restrict__ Whh,
                          const float* __restrict__ bih,
                          const float* __restrict__ bhh,
                          unsigned short* __restrict__ whhPack,
                          unsigned short* __restrict__ wihPack,
                          float* __restrict__ biasAll) {
    const int stride = gridDim.x * blockDim.x;
    const int idx    = blockIdx.x * blockDim.x + threadIdx.x;

    for (int p = idx; p < LL * PACK_PER_LAYER; p += stride) {
        int l = p / PACK_PER_LAYER, r = p % PACK_PER_LAYER;
        int f = r >> 9, li = r & 511;
        int lane = li >> 4, i = li & 15;
        int nt = f & 15, kt = f >> 4;
        int n = nt * 16 + (lane & 15);
        int k = kt * 32 + (lane >> 4) * 8 + (i & 7) + ((i >> 3) << 4);
        const float* W = (l == 0) ? Whh0 : (Whh + (size_t)(l - 1) * 256 * HH);
        whhPack[p] = f2bf(W[n * HH + k]);
    }
    for (int p = idx; p < (LL - 1) * PACK_PER_LAYER; p += stride) {
        int l = p / PACK_PER_LAYER, r = p % PACK_PER_LAYER;
        int f = r >> 9, li = r & 511;
        int lane = li >> 4, i = li & 15;
        int nt = f & 15, kt = f >> 4;
        int n = nt * 16 + (lane & 15);
        int k = kt * 32 + (lane >> 4) * 8 + (i & 7) + ((i >> 3) << 4);
        wihPack[p] = f2bf(Wih[(size_t)l * 256 * HH + n * HH + k]);
    }
    for (int p = idx; p < LL * 256; p += stride) {
        int l = p >> 8, n = p & 255;
        biasAll[p] = (l == 0) ? (bih0[n] + bhh0[n])
                              : (bih[(l - 1) * 256 + n] + bhh[(l - 1) * 256 + n]);
    }
}

// ---------------- main persistent kernel: whole encoder + decoder per batch slice --------
__global__ __launch_bounds__(256, 1)
void lstm_main(const float* __restrict__ x,
               const float* __restrict__ wih0,     // (256,1)
               const float* __restrict__ fcw,
               const float* __restrict__ fcb,
               const unsigned short* __restrict__ whhPack,
               const unsigned short* __restrict__ wihPack,
               const float* __restrict__ biasAll,
               unsigned short* __restrict__ seq,   // [blk][t][32][64] bf16, in-place per layer
               float* __restrict__ cstate,         // fragment-packed f32 per (blk,layer,wave)
               float* __restrict__ out) {
    __shared__ alignas(16) unsigned short hstate[LL][ROWS * HH];  // decode h-state (bf16)
    __shared__ alignas(16) unsigned short bufA[ROWS * HH];
    __shared__ alignas(16) unsigned short bufB[ROWS * HH];
    __shared__ float biasS[LL * 256];
    __shared__ float wih0S[256];
    __shared__ float fcwS[HH];
    __shared__ float ybuf[ROWS];
    __shared__ float fcbS;

    const int tid  = threadIdx.x;
    const int lane = tid & 31, w = tid >> 5;
    const int m = w >> 2, cg = w & 3;
    const int g = lane >> 4, ln = lane & 15;
    const int blk = blockIdx.x;
    const int b0  = blk * ROWS;

    for (int i = tid; i < LL * 256; i += 256) biasS[i] = biasAll[i];
    wih0S[tid] = wih0[tid];
    if (tid < HH) fcwS[tid] = fcw[tid];
    if (tid == 0) fcbS = fcb[0];
    __syncthreads();

    float c8[8];

    // ==================== ENCODER ====================
    // ---- layer 0 (scalar input, rank-1 Wih path folded into accumulator init) ----
    {
        for (int i = tid; i < ROWS * HH; i += 256) bufB[i] = 0;
#pragma unroll
        for (int j = 0; j < 8; ++j) c8[j] = 0.f;
        const unsigned short* whh = whhPack;
        for (int t = 0; t < TT; ++t) {
            if (tid < ROWS) ybuf[tid] = x[(size_t)(b0 + tid) * TT + t];
            __syncthreads();
            v16bf ah0 = load_afrag(bufB, m, 0, lane);
            v16bf ah1 = load_afrag(bufB, m, 1, lane);
            v8f acc[4];
#pragma unroll
            for (int gi = 0; gi < 4; ++gi) {
                int nt = gi * 4 + cg;
                float bv = biasS[nt * 16 + ln];
                float wv = wih0S[nt * 16 + ln];
                v8f a;
#pragma unroll
                for (int j = 0; j < 8; ++j) {
                    int row = 16 * m + j + 8 * g;
                    a[j] = bv + ybuf[row] * wv;
                }
                a = wmma_bf(ah0, load_bfrag(whh, nt, lane), a);
                a = wmma_bf(ah1, load_bfrag(whh, 16 + nt, lane), a);
                acc[gi] = a;
            }
            __syncthreads();
            cell_update(acc, c8, bufB, m, cg, g, ln);
            __syncthreads();
            ((uint4*)(seq + ((size_t)(blk * TT + t)) * (ROWS * HH)))[tid] = ((const uint4*)bufB)[tid];
        }
        float* cp = cstate + (((size_t)blk * LL + 0) * 8 + w) * 256 + lane * 8;
#pragma unroll
        for (int j = 0; j < 8; ++j) cp[j] = c8[j];
        for (int i = tid; i < ROWS * HH; i += 256) hstate[0][i] = bufB[i];
        __syncthreads();
    }

    // ---- layers 1..14 ----
    for (int l = 1; l < LL; ++l) {
        const unsigned short* whh = whhPack + (size_t)l * PACK_PER_LAYER;
        const unsigned short* wih = wihPack + (size_t)(l - 1) * PACK_PER_LAYER;
        const float* bl = biasS + l * 256;
        for (int i = tid; i < ROWS * HH; i += 256) bufB[i] = 0;
#pragma unroll
        for (int j = 0; j < 8; ++j) c8[j] = 0.f;
        for (int t = 0; t < TT; ++t) {
            ((uint4*)bufA)[tid] = ((const uint4*)(seq + ((size_t)(blk * TT + t)) * (ROWS * HH)))[tid];
            __syncthreads();
            v16bf ax0 = load_afrag(bufA, m, 0, lane);
            v16bf ax1 = load_afrag(bufA, m, 1, lane);
            v16bf ah0 = load_afrag(bufB, m, 0, lane);
            v16bf ah1 = load_afrag(bufB, m, 1, lane);
            v8f acc[4];
#pragma unroll
            for (int gi = 0; gi < 4; ++gi) {
                int nt = gi * 4 + cg;
                float bv = bl[nt * 16 + ln];
                v8f a = {bv, bv, bv, bv, bv, bv, bv, bv};
                a = wmma_bf(ax0, load_bfrag(wih, nt, lane), a);
                a = wmma_bf(ax1, load_bfrag(wih, 16 + nt, lane), a);
                a = wmma_bf(ah0, load_bfrag(whh, nt, lane), a);
                a = wmma_bf(ah1, load_bfrag(whh, 16 + nt, lane), a);
                acc[gi] = a;
            }
            __syncthreads();
            cell_update(acc, c8, bufB, m, cg, g, ln);
            __syncthreads();
            ((uint4*)(seq + ((size_t)(blk * TT + t)) * (ROWS * HH)))[tid] = ((const uint4*)bufB)[tid];
        }
        float* cp = cstate + (((size_t)blk * LL + l) * 8 + w) * 256 + lane * 8;
#pragma unroll
        for (int j = 0; j < 8; ++j) cp[j] = c8[j];
        for (int i = tid; i < ROWS * HH; i += 256) hstate[l][i] = bufB[i];
        __syncthreads();
    }

    // ==================== DECODER ====================
    if (tid < ROWS) ybuf[tid] = x[(size_t)(b0 + tid) * TT + (TT - 1)];
    __syncthreads();

    for (int t = 0; t < TT; ++t) {
        // ---- layer 0 (scalar input y) ----
        {
            const unsigned short* whh = whhPack;
            float* cp = cstate + (((size_t)blk * LL + 0) * 8 + w) * 256 + lane * 8;
#pragma unroll
            for (int j = 0; j < 8; ++j) c8[j] = cp[j];
            v16bf ah0 = load_afrag(hstate[0], m, 0, lane);
            v16bf ah1 = load_afrag(hstate[0], m, 1, lane);
            v8f acc[4];
#pragma unroll
            for (int gi = 0; gi < 4; ++gi) {
                int nt = gi * 4 + cg;
                float bv = biasS[nt * 16 + ln];
                float wv = wih0S[nt * 16 + ln];
                v8f a;
#pragma unroll
                for (int j = 0; j < 8; ++j) {
                    int row = 16 * m + j + 8 * g;
                    a[j] = bv + ybuf[row] * wv;
                }
                a = wmma_bf(ah0, load_bfrag(whh, nt, lane), a);
                a = wmma_bf(ah1, load_bfrag(whh, 16 + nt, lane), a);
                acc[gi] = a;
            }
            cell_update(acc, c8, bufA, m, cg, g, ln);
#pragma unroll
            for (int j = 0; j < 8; ++j) cp[j] = c8[j];
            __syncthreads();
            for (int i = tid; i < ROWS * HH; i += 256) hstate[0][i] = bufA[i];
            __syncthreads();
        }
        // ---- layers 1..14 (ping-pong bufA/bufB) ----
        for (int l = 1; l < LL; ++l) {
            const unsigned short* whh = whhPack + (size_t)l * PACK_PER_LAYER;
            const unsigned short* wih = wihPack + (size_t)(l - 1) * PACK_PER_LAYER;
            const float* bl = biasS + l * 256;
            unsigned short* xin  = ((l - 1) & 1) ? bufB : bufA;
            unsigned short* xout = (l & 1) ? bufB : bufA;
            float* cp = cstate + (((size_t)blk * LL + l) * 8 + w) * 256 + lane * 8;
#pragma unroll
            for (int j = 0; j < 8; ++j) c8[j] = cp[j];
            v16bf ax0 = load_afrag(xin, m, 0, lane);
            v16bf ax1 = load_afrag(xin, m, 1, lane);
            v16bf ah0 = load_afrag(hstate[l], m, 0, lane);
            v16bf ah1 = load_afrag(hstate[l], m, 1, lane);
            v8f acc[4];
#pragma unroll
            for (int gi = 0; gi < 4; ++gi) {
                int nt = gi * 4 + cg;
                float bv = bl[nt * 16 + ln];
                v8f a = {bv, bv, bv, bv, bv, bv, bv, bv};
                a = wmma_bf(ax0, load_bfrag(wih, nt, lane), a);
                a = wmma_bf(ax1, load_bfrag(wih, 16 + nt, lane), a);
                a = wmma_bf(ah0, load_bfrag(whh, nt, lane), a);
                a = wmma_bf(ah1, load_bfrag(whh, 16 + nt, lane), a);
                acc[gi] = a;
            }
            cell_update(acc, c8, xout, m, cg, g, ln);
#pragma unroll
            for (int j = 0; j < 8; ++j) cp[j] = c8[j];
            __syncthreads();
            for (int i = tid; i < ROWS * HH; i += 256) hstate[l][i] = xout[i];
            __syncthreads();
        }
        // ---- projection: y = h14 @ fcw + fcb ; h14 lives in bufA (layer 14 output) ----
        if (tid < ROWS) {
            float s = fcbS;
#pragma unroll 8
            for (int k = 0; k < HH; ++k) s += bf2f(bufA[tid * HH + k]) * fcwS[k];
            ybuf[tid] = s;
            out[(size_t)(b0 + tid) * TT + t] = s;  // preds.T -> (B,T) row-major
        }
        __syncthreads();
    }
}

extern "C" void kernel_launch(void* const* d_in, const int* in_sizes, int n_in,
                              void* d_out, int out_size, void* d_ws, size_t ws_size,
                              hipStream_t stream) {
    (void)in_sizes; (void)n_in; (void)out_size; (void)ws_size;

    const float* x    = (const float*)d_in[0];
    const float* Wih0 = (const float*)d_in[1];
    const float* Whh0 = (const float*)d_in[2];
    const float* bih0 = (const float*)d_in[3];
    const float* bhh0 = (const float*)d_in[4];
    const float* Wih  = (const float*)d_in[5];
    const float* Whh  = (const float*)d_in[6];
    const float* bih  = (const float*)d_in[7];
    const float* bhh  = (const float*)d_in[8];
    const float* fcw  = (const float*)d_in[9];
    const float* fcb  = (const float*)d_in[10];

    char* ws = (char*)d_ws;
    unsigned short* whhPack = (unsigned short*)(ws + 0);                       // 491520 B
    unsigned short* wihPack = (unsigned short*)(ws + 491520);                  // 458752 B
    float*          biasAll = (float*)(ws + 950272);                           //  15360 B
    unsigned short* seq     = (unsigned short*)(ws + (1u << 20));              // 20 MB
    float*          cstate  = (float*)(ws + (1u << 20) + 20971520u);           // 30 MB

    lstm_prep<<<256, 256, 0, stream>>>(Whh0, bih0, bhh0, Wih, Whh, bih, bhh,
                                       whhPack, wihPack, biasAll);
    lstm_main<<<NBLK, 256, 0, stream>>>(x, Wih0, fcw, fcb, whhPack, wihPack, biasAll,
                                        seq, cstate, (float*)d_out);
}
